// RGCNLayer_52905407152976
// MI455X (gfx1250) — matrix operand order repaired and checked by reading, not compiled, not run on hardware
//
#include <hip/hip_runtime.h>
#include <hip/hip_bf16.h>

// Problem constants (match reference)
#define NNODES 12500
#define NEDGES 200000
#define DIN    32
#define DOUT   32
#define DATTN  32
#define NREL   200
#define NBASES 50
#define DCAT   128   // 2*IN + 2*ATTN

typedef __attribute__((ext_vector_type(16))) __bf16 v16bf;
typedef __attribute__((ext_vector_type(8)))  float  v8f;
typedef __attribute__((ext_vector_type(4)))  unsigned int u32x4;
typedef __attribute__((ext_vector_type(8)))  int  i32x8;
typedef __attribute__((ext_vector_type(4)))  int  i32x4;

// CDNA5 16-bit A-matrix 16x32 layout: lane L holds row M=L%16.
// VGPR v (element pair j=2v,2v+1): lanes 0-15: v0..3 -> K=0..7, v4..7 -> K=16..23
//                                  lanes16-31: v0..3 -> K=8..15, v4..7 -> K=24..31
__device__ __forceinline__ int a_klocal(int j, int lh) {
    int v = j >> 1, h = j & 1;
    int base = ((v < 4) ? 0 : 16) + (lh ? 8 : 0);
    return base + 2 * (v & 3) + h;
}
// B-matrix 32x16: N = lane%16; same K order as A (lanes 0-15: K=0..7,16..23;
// lanes 16-31: K=8..15,24..31)
__device__ __forceinline__ int b_klocal(int j, int lh) {
    return a_klocal(j, lh);
}

// Tensor Data Mover: DMA a contiguous [16 rows x 32 f32] tile (2 KB) into LDS.
// D# per CDNA5 ISA 8.3/8.4: group0 = {count/flags, lds_addr, global_addr, type=2},
// group1 = {data_size=4B, tensor_dim0=32, tensor_dim1=16, tile 32x16, stride0=32}.
// This toolchain's builtin is the 6-arg clang-23 form:
//   (u32x4 g0, i32x8 g1, i32x4 g2, i32x4 g3, i32x8 extra, i32 cpol)
__device__ __forceinline__ void tdm_load_tile_16x32_f32(const float* gsrc,
                                                        unsigned lds_off) {
    unsigned long long ga = (unsigned long long)(uintptr_t)gsrc;
    u32x4 g0;
    g0[0] = 1u;                                          // count=1, user descriptor
    g0[1] = lds_off;                                     // lds_addr (bytes)
    g0[2] = (unsigned)(ga & 0xFFFFFFFFu);                // global_addr[31:0]
    g0[3] = (unsigned)((ga >> 32) & 0x01FFFFFFu)         // global_addr[56:32]
          | (2u << 30);                                  // type=2 ("image")
    i32x8 g1;
    g1[0] = (int)(2u << 16);        // workgroup_mask=0, data_size=2 (4 bytes)
    g1[1] = (int)(32u << 16);       // tensor_dim0 = 32 (bits 63:48)
    g1[2] = (int)(16u << 16);       // tensor_dim0 hi=0, tensor_dim1 = 16 (bits 95:80)
    g1[3] = (int)(32u << 16);       // tensor_dim1 hi=0, tile_dim0 = 32 (bits 127:112)
    g1[4] = (int)16u;               // tile_dim1 = 16, tile_dim2 = 0
    g1[5] = (int)32u;               // tensor_dim0_stride = 32 (bits 191:160)
    g1[6] = 0;                      // stride0 hi, tensor_dim1_stride lo
    g1[7] = 0;
    i32x4 z4 = {0, 0, 0, 0};        // groups 2/3 unused (<=2D tensor)
    i32x8 z8 = {0, 0, 0, 0, 0, 0, 0, 0};
    __builtin_amdgcn_tensor_load_to_lds(g0, g1, z4, z4, z8, 0);
}

__global__ void zero_kernel(float* __restrict__ p, int n) {
    int i = blockIdx.x * blockDim.x + threadIdx.x;
    if (i < n) p[i] = 0.f;
}

// rel_w[r,i,o] = sum_b w_comp[r,b] * weight[b,i,o]   -> d_ws  (200 x 32 x 32 f32)
__global__ void __launch_bounds__(256)
basis_kernel(const float* __restrict__ w_comp,
             const float* __restrict__ weight,
             float* __restrict__ rel_w) {
    int r = blockIdx.x;
    int t = threadIdx.x;              // 256 threads, 4 elems each (32*32=1024)
    float acc0 = 0.f, acc1 = 0.f, acc2 = 0.f, acc3 = 0.f;
    for (int b = 0; b < NBASES; ++b) {
        float w = w_comp[r * NBASES + b];
        const float4 wb = *(const float4*)(weight + (long)b * (DIN * DOUT) + t * 4);
        acc0 = fmaf(w, wb.x, acc0);
        acc1 = fmaf(w, wb.y, acc1);
        acc2 = fmaf(w, wb.z, acc2);
        acc3 = fmaf(w, wb.w, acc3);
    }
    float4 o; o.x = acc0; o.y = acc1; o.z = acc2; o.w = acc3;
    *(float4*)(rel_w + (long)r * (DIN * DOUT) + t * 4) = o;
}

// One wave32 per 16-edge tile; 8 waves (256 threads) per block.
__global__ void __launch_bounds__(256)
rgcn_edge_kernel(const float* __restrict__ node_feat,   // [N,32]
                 const float* __restrict__ tgt_rel,     // [E,32] total_target_relation
                 const float* __restrict__ rel_embed,   // [E,32] total_relation_embed
                 const float* __restrict__ slw,         // [32,32] self_loop_weight
                 const float* __restrict__ A_w,         // [32,128]
                 const float* __restrict__ A_b,         // [32]
                 const float* __restrict__ B_w,         // [32]
                 const float* __restrict__ B_b,         // [1]
                 const int*   __restrict__ edge,        // [2,E]
                 const int*   __restrict__ rel,         // [E]
                 const float* __restrict__ rel_w,       // [200,32,32] (ws)
                 float*       __restrict__ out)         // [N,32]
{
    __shared__ float relEmbT[8][16][DATTN];     // TDM landing pads (16 KB each)
    __shared__ float tgtRelT[8][16][DATTN];
    __shared__ float msgTile[8][16][DOUT + 1];  // +1 pad: avoid bank conflicts
    __shared__ float aVals[8][16];
    __shared__ int   tIdxS[8][16];

    const int wave = threadIdx.x >> 5;
    const int lane = threadIdx.x & 31;
    const int lh   = lane >> 4;     // lane half (K-half for A/B, M-half for C/D)
    const int lm   = lane & 15;     // row / column within tile
    const long tile = (long)blockIdx.x * 8 + wave;
    const bool valid = (tile * 16) < NEDGES;    // uniform per wave -> EXEC all-1s at WMMA

    v8f accA0 = {}; v8f accA1 = {};   // attention H[16,32], two 16-wide N tiles
    v8f accS0 = {}; v8f accS1 = {};   // self-loop curr[16,32]

    int e = valid ? (int)(tile * 16 + lm) : 0;
    const int sIdx = edge[e];
    const int tIdx = edge[NEDGES + e];
    const int rr   = rel[e];
    const float* __restrict__ srcRow  = node_feat + (long)sIdx * DIN;
    const float* __restrict__ tgtRow  = node_feat + (long)tIdx * DIN;
    const float* __restrict__ relwRow = rel_w + (long)rr * (DIN * DOUT);

    if (valid) {
        // ---- async: DMA the two HBM-resident embedding tiles into LDS ----
        const long tb = tile * 16 * DATTN;      // contiguous 16x32 f32 tile base
        tdm_load_tile_16x32_f32(rel_embed + tb,
                                (unsigned)(uintptr_t)&relEmbT[wave][0][0]);
        tdm_load_tile_16x32_f32(tgt_rel + tb,
                                (unsigned)(uintptr_t)&tgtRelT[wave][0][0]);

        __builtin_prefetch(relwRow, 0, 1);       // global_prefetch_b8 (L2-hot rel_w)
        if (lh == 0) tIdxS[wave][lm] = tIdx;

        // ---- msg: per-edge relation-specific matvec (f32 exact, overlaps TDM) ----
        {
            float msgv[16];
            #pragma unroll
            for (int o = 0; o < 16; ++o) msgv[o] = 0.f;
            for (int i = 0; i < DIN; ++i) {
                float s = srcRow[i];
                const float4* __restrict__ w4 =
                    (const float4*)(relwRow + i * DOUT + lh * 16);
                #pragma unroll
                for (int u = 0; u < 4; ++u) {
                    float4 w = w4[u];
                    msgv[4*u+0] = fmaf(s, w.x, msgv[4*u+0]);
                    msgv[4*u+1] = fmaf(s, w.y, msgv[4*u+1]);
                    msgv[4*u+2] = fmaf(s, w.z, msgv[4*u+2]);
                    msgv[4*u+3] = fmaf(s, w.w, msgv[4*u+3]);
                }
            }
            #pragma unroll
            for (int o = 0; o < 16; ++o)
                msgTile[wave][lm][lh * 16 + o] = msgv[o];  // lane: edge lm, 16 outs
        }

        // ---- self-loop: curr = tgt[16,32] @ slw[32,32]  (one K-chunk, 2 N-tiles) ----
        {
            v16bf af, b0, b1;
            #pragma unroll
            for (int v = 0; v < 8; ++v) {       // A rows: consecutive K pairs -> float2
                float2 x = *(const float2*)(tgtRow + a_klocal(2 * v, lh));
                af[2*v]   = (__bf16)x.x;
                af[2*v+1] = (__bf16)x.y;
            }
            #pragma unroll
            for (int j = 0; j < 16; ++j) {      // B: K-major stride 32, scalar loads
                int kb = b_klocal(j, lh);
                b0[j] = (__bf16)slw[kb * DOUT + lm];
                b1[j] = (__bf16)slw[kb * DOUT + 16 + lm];
            }
            accS0 = __builtin_amdgcn_wmma_f32_16x16x32_bf16(
                        false, af, false, b0, (short)0, accS0, false, false);
            accS1 = __builtin_amdgcn_wmma_f32_16x16x32_bf16(
                        false, af, false, b1, (short)0, accS1, false, false);
        }

        // ---- wait for TDM tiles, then attention GEMM: H = Ecat[16,128] @ A_w^T ----
        __builtin_amdgcn_s_wait_tensorcnt(0);
        {
            const float* catSrc[4] = { srcRow, tgtRow,
                                       &relEmbT[wave][lm][0],
                                       &tgtRelT[wave][lm][0] };
            #pragma unroll
            for (int kc = 0; kc < 4; ++kc) {
                const float* __restrict__ abase = catSrc[kc];
                v16bf af, b0, b1;
                #pragma unroll
                for (int v = 0; v < 8; ++v) {
                    float2 x = *(const float2*)(abase + a_klocal(2 * v, lh));
                    af[2*v]   = (__bf16)x.x;
                    af[2*v+1] = (__bf16)x.y;
                }
                #pragma unroll
                for (int v = 0; v < 8; ++v) {   // b_klocal consecutive in j -> float2
                    int kg = kc * 32 + b_klocal(2 * v, lh);
                    float2 y0 = *(const float2*)(A_w + lm * DCAT + kg);
                    float2 y1 = *(const float2*)(A_w + (16 + lm) * DCAT + kg);
                    b0[2*v] = (__bf16)y0.x;  b0[2*v+1] = (__bf16)y0.y;
                    b1[2*v] = (__bf16)y1.x;  b1[2*v+1] = (__bf16)y1.y;
                }
                accA0 = __builtin_amdgcn_wmma_f32_16x16x32_bf16(
                            false, af, false, b0, (short)0, accA0, false, false);
                accA1 = __builtin_amdgcn_wmma_f32_16x16x32_bf16(
                            false, af, false, b1, (short)0, accA1, false, false);
            }
        }

        // ---- a[m] = sigmoid(B_w . relu(H[m,:] + A_b) + B_b)  (in-half reduction) ----
        {
            float bw0 = B_w[lm], bw1 = B_w[16 + lm];
            float ab0 = A_b[lm], ab1 = A_b[16 + lm];
            float part[8];
            #pragma unroll
            for (int j = 0; j < 8; ++j) {
                float h0 = fmaxf(accA0[j] + ab0, 0.f);
                float h1 = fmaxf(accA1[j] + ab1, 0.f);
                part[j] = h0 * bw0 + h1 * bw1;
            }
            #pragma unroll
            for (int m = 1; m < 16; m <<= 1) {
                #pragma unroll
                for (int j = 0; j < 8; ++j)
                    part[j] += __shfl_xor(part[j], m, 32);
            }
            if (lm == 0) {
                float bb = B_b[0];
                #pragma unroll
                for (int j = 0; j < 8; ++j)   // rows 0-7 in half0, 8-15 in half1
                    aVals[wave][j + 8 * lh] = 1.f / (1.f + __expf(-(part[j] + bb)));
            }
        }
    }

    __syncthreads();  // msgTile / aVals / tIdxS visible (cross-lane LDS)

    if (valid) {
        // C/D layout: VGPR j -> row M = j + 8*lh, col n = lm (acc*0) / 16+lm (acc*1)
        #pragma unroll
        for (int j = 0; j < 8; ++j) {
            int M = j + 8 * lh;
            float a = aVals[wave][M];
            int tn = tIdxS[wave][M];
            float v0 = accS0[j] + msgTile[wave][M][lm] * a;
            float v1 = accS1[j] + msgTile[wave][M][16 + lm] * a;
            atomicAdd(out + (long)tn * DOUT + lm,      v0);
            atomicAdd(out + (long)tn * DOUT + 16 + lm, v1);
        }
    }
}

extern "C" void kernel_launch(void* const* d_in, const int* in_sizes, int n_in,
                              void* d_out, int out_size, void* d_ws, size_t ws_size,
                              hipStream_t stream) {
    const float* node_feat = (const float*)d_in[0];
    const float* tgt_rel   = (const float*)d_in[1];
    const float* rel_embed = (const float*)d_in[2];
    const float* weight    = (const float*)d_in[3];
    const float* w_comp    = (const float*)d_in[4];
    const float* slw       = (const float*)d_in[5];
    const float* A_w       = (const float*)d_in[6];
    const float* A_b       = (const float*)d_in[7];
    const float* B_w       = (const float*)d_in[8];
    const float* B_b       = (const float*)d_in[9];
    const int*   edge      = (const int*)d_in[10];
    const int*   rel       = (const int*)d_in[11];
    float* out   = (float*)d_out;
    float* rel_w = (float*)d_ws;   // 200*32*32 f32 = 800 KB scratch

    zero_kernel<<<(NNODES * DOUT + 255) / 256, 256, 0, stream>>>(out, NNODES * DOUT);
    basis_kernel<<<NREL, 256, 0, stream>>>(w_comp, weight, rel_w);

    const int tiles  = (NEDGES + 15) / 16;   // 12500 (exact)
    const int blocks = (tiles + 7) / 8;      // 1563
    rgcn_edge_kernel<<<blocks, 256, 0, stream>>>(node_feat, tgt_rel, rel_embed,
                                                 slw, A_w, A_b, B_w, B_b,
                                                 edge, rel, rel_w, out);
}
